// GIN_39247411151131
// MI455X (gfx1250) — matrix operand order repaired and checked by reading, not compiled, not run on hardware
//
#include <hip/hip_runtime.h>

#define N_NODES 50000
#define E_EDGES 800000
#define DIM     200      // embed dim (K)
#define NCLS    10       // classes
#define KPAD    224      // 7 * 32, padded K for WMMA
#define KSTEPS  7        // K iterations of 32

typedef __attribute__((ext_vector_type(16))) _Float16 v16h;
typedef __attribute__((ext_vector_type(8)))  float    v8f;

// ---------------------------------------------------------------------------
// h[i] = 0.1*(1+eps0) * W0[i]   (self term; also serves as the zero-init)
// ---------------------------------------------------------------------------
__global__ void init_h_kernel(const float* __restrict__ W0,
                              const float* __restrict__ eps0,
                              float* __restrict__ h) {
    const float s = 0.1f * (1.0f + eps0[0]);
    const int total4 = (N_NODES * DIM) / 4;   // 2.5M float4
    const float4* __restrict__ w4 = (const float4*)W0;
    float4* __restrict__ h4 = (float4*)h;
    for (int i = blockIdx.x * blockDim.x + threadIdx.x; i < total4;
         i += gridDim.x * blockDim.x) {
        float4 w = w4[i];
        h4[i] = make_float4(s * w.x, s * w.y, s * w.z, s * w.w);
    }
}

// ---------------------------------------------------------------------------
// COO SpMM scatter: dst[r,:] += v * src[c,:], one wave32 per edge.
// 32 lanes sweep the 200-float row -> coalesced reads + f32 global atomics
// that stay resident in the 192MB L2 (src/dst are 40MB each).
// ---------------------------------------------------------------------------
__global__ void spmm_scatter_kernel(const int*   __restrict__ rows,
                                    const int*   __restrict__ cols,
                                    const float* __restrict__ vals,
                                    const float* __restrict__ src,
                                    float*       __restrict__ dst,
                                    int nedges) {
    const int lane   = threadIdx.x & 31;
    const int warp   = (blockIdx.x * blockDim.x + threadIdx.x) >> 5;
    const int nwarps = (gridDim.x * blockDim.x) >> 5;
    for (int e = warp; e < nedges; e += nwarps) {
        const int   r = rows[e];
        const int   c = cols[e];
        const float v = vals[e];
        const float* __restrict__ srow = src + (size_t)c * DIM;
        float*       __restrict__ drow = dst + (size_t)r * DIM;
        for (int d = lane; d < DIM; d += 32) {
            atomicAdd(&drow[d], v * srow[d]);
        }
    }
}

// ---------------------------------------------------------------------------
// h = relu(h);  ax = 0
// ---------------------------------------------------------------------------
__global__ void relu_zero_kernel(float* __restrict__ h, float* __restrict__ ax) {
    const int total4 = (N_NODES * DIM) / 4;
    float4* __restrict__ h4  = (float4*)h;
    float4* __restrict__ ax4 = (float4*)ax;
    const float4 z = make_float4(0.f, 0.f, 0.f, 0.f);
    for (int i = blockIdx.x * blockDim.x + threadIdx.x; i < total4;
         i += gridDim.x * blockDim.x) {
        float4 v = h4[i];
        v.x = fmaxf(v.x, 0.f); v.y = fmaxf(v.y, 0.f);
        v.z = fmaxf(v.z, 0.f); v.w = fmaxf(v.w, 0.f);
        h4[i]  = v;
        ax4[i] = z;
    }
}

// ---------------------------------------------------------------------------
// out[N,10] = (ax + 0.1*(1+eps1)*h) @ W1   via v_wmma_f32_16x16x32_f16.
// One wave32 computes a 16(M=nodes) x 16(N=classes, 10 used) tile;
// K = 200 padded to 7*32. W1 staged in LDS as [n][kpad] halves so each
// B fragment K-pair is one 32-bit ds load.
// ---------------------------------------------------------------------------
__global__ __launch_bounds__(256) void gemm_wmma_kernel(
        const float* __restrict__ ax,
        const float* __restrict__ h,
        const float* __restrict__ W1,
        const float* __restrict__ eps1,
        float* __restrict__ out) {
    __shared__ _Float16 Blds[16 * KPAD];   // 7168 bytes

    // Cooperative stage of W1 (200x10, row-major) into padded [n][k] layout.
    for (int idx = threadIdx.x; idx < 16 * KPAD; idx += blockDim.x) {
        const int n = idx / KPAD;
        const int k = idx % KPAD;
        const float v = (n < NCLS && k < DIM) ? W1[k * NCLS + n] : 0.0f;
        Blds[idx] = (_Float16)v;
    }
    __syncthreads();

    const float s1 = 0.1f * (1.0f + eps1[0]);

    const int lane  = threadIdx.x & 31;
    const int wid   = threadIdx.x >> 5;
    const int wpb   = blockDim.x >> 5;                 // 8 waves / block
    const int gwarp = blockIdx.x * wpb + wid;
    const int nwarp = gridDim.x * wpb;
    const int half  = lane >> 4;                       // 0: lanes 0-15, 1: 16-31
    const int mn    = lane & 15;                       // A row / B,C,D column

    const int ntiles = N_NODES / 16;                   // 3125 exactly
    for (int tile = gwarp; tile < ntiles; tile += nwarp) {
        const int row = tile * 16 + mn;                // node row for A fragment
        const float* __restrict__ arow = ax + (size_t)row * DIM;
        const float* __restrict__ hrow = h  + (size_t)row * DIM;

        v8f acc = {};
        #pragma unroll
        for (int t = 0; t < KSTEPS; ++t) {
            const int kbaseA = t * 32 + half * 8;      // ISA 16-bit A layout
            v16h A, B;
            #pragma unroll
            for (int g = 0; g < 8; ++g) {
                // A fragment: VGPR g holds K pair {kA, kA+1}
                const int kA = kbaseA + (g < 4 ? 2 * g : 16 + 2 * (g - 4));
                float zx = 0.f, zy = 0.f;
                if (kA < DIM) {                        // kA even -> kA+1 < DIM too
                    const float2 a  = *(const float2*)(arow + kA);
                    const float2 hh = *(const float2*)(hrow + kA);
                    zx = fmaf(s1, hh.x, a.x);
                    zy = fmaf(s1, hh.y, a.y);
                }
                A[2 * g]     = (_Float16)zx;
                A[2 * g + 1] = (_Float16)zy;

                // B fragment: same K order as A (lanes 0-15 K=0..7,16..23; lanes 16-31 K=8..15,24..31)
                const int kB = t * 32 + half * 8 + (g < 4 ? 2 * g : 16 + 2 * (g - 4));
                const _Float16* bp = &Blds[mn * KPAD + kB];
                B[2 * g]     = bp[0];
                B[2 * g + 1] = bp[1];
            }
            acc = __builtin_amdgcn_wmma_f32_16x16x32_f16(
                      /*neg_a=*/false, A, /*neg_b=*/false, B,
                      /*c_mod=*/(short)0, acc,
                      /*reuse_a=*/false, /*reuse_b=*/false);
        }

        // D layout: lane column n = mn; VGPR g holds row m = g + 8*half.
        if (mn < NCLS) {
            #pragma unroll
            for (int g = 0; g < 8; ++g) {
                const int m = g + 8 * half;
                out[(size_t)(tile * 16 + m) * NCLS + mn] = acc[g];
            }
        }
    }
}

// ---------------------------------------------------------------------------
// Inputs (setup_inputs order):
//  0:x  1:rows0  2:cols0  3:vals0  4:rows1  5:cols1  6:vals1
//  7:W0 [N*D]    8:W1 [D*C]        9:eps0   10:eps1
// Workspace: h [N*D] f32 (40MB) + ax [N*D] f32 (40MB) = 80MB.
// ---------------------------------------------------------------------------
extern "C" void kernel_launch(void* const* d_in, const int* in_sizes, int n_in,
                              void* d_out, int out_size, void* d_ws, size_t ws_size,
                              hipStream_t stream) {
    const int*   rows0 = (const int*)  d_in[1];
    const int*   cols0 = (const int*)  d_in[2];
    const float* vals0 = (const float*)d_in[3];
    const int*   rows1 = (const int*)  d_in[4];
    const int*   cols1 = (const int*)  d_in[5];
    const float* vals1 = (const float*)d_in[6];
    const float* W0    = (const float*)d_in[7];
    const float* W1    = (const float*)d_in[8];
    const float* eps0  = (const float*)d_in[9];
    const float* eps1  = (const float*)d_in[10];
    float* out = (float*)d_out;

    float* h  = (float*)d_ws;
    float* ax = h + (size_t)N_NODES * DIM;

    // Layer 0: h = A0@W0 + A1@W0 + 0.1*(1+eps0)*W0 ; h = relu(h); ax = 0
    init_h_kernel<<<2048, 256, 0, stream>>>(W0, eps0, h);
    spmm_scatter_kernel<<<8192, 256, 0, stream>>>(rows0, cols0, vals0, W0, h, E_EDGES);
    spmm_scatter_kernel<<<8192, 256, 0, stream>>>(rows1, cols1, vals1, W0, h, E_EDGES);
    relu_zero_kernel<<<2048, 256, 0, stream>>>(h, ax);

    // Layer 1: ax = A0@h + A1@h ; out = (ax + 0.1*(1+eps1)*h) @ W1
    spmm_scatter_kernel<<<8192, 256, 0, stream>>>(rows0, cols0, vals0, h, ax, E_EDGES);
    spmm_scatter_kernel<<<8192, 256, 0, stream>>>(rows1, cols1, vals1, h, ax, E_EDGES);
    gemm_wmma_kernel<<<(N_NODES / 16 + 7) / 8, 256, 0, stream>>>(ax, h, W1, eps1, out);
}
